// MambaBlock_19275813224604
// MI455X (gfx1250) — matrix operand order repaired and checked by reading, not compiled, not run on hardware
//
#include <hip/hip_runtime.h>
#include <hip/hip_bf16.h>

#define SEQ      2048
#define D_MODEL  768
#define D_INNER  1536
#define D_STATE  16
#define DT_RANK  48
#define XDBL_W   (DT_RANK + 2 * D_STATE)   // 80

typedef __attribute__((ext_vector_type(16))) __bf16 v16bf;
typedef __attribute__((ext_vector_type(8)))  __bf16 v8bf;
typedef __attribute__((ext_vector_type(8)))  float  v8f;

// ---------------------------------------------------------------------------
// fp32 -> bf16 conversion (one-time weight/activation downconvert)
// ---------------------------------------------------------------------------
__global__ void cvt_f32_bf16(const float* __restrict__ in, __bf16* __restrict__ out, long n) {
  long i = (long)blockIdx.x * blockDim.x + threadIdx.x;
  if (i < n) out[i] = (__bf16)in[i];
}

// dt_proj_w (1536,48) -> transposed (48,1536) fp32, so the dt GEMV is coalesced
__global__ void transpose_dtw(const float* __restrict__ in, float* __restrict__ out) {
  long i = (long)blockIdx.x * blockDim.x + threadIdx.x;
  if (i >= (long)D_INNER * DT_RANK) return;
  int c = (int)(i / DT_RANK), j = (int)(i % DT_RANK);
  out[(long)j * D_INNER + c] = in[i];
}

// ---------------------------------------------------------------------------
// Register-blocked bf16 WMMA GEMM:
//   C[M=2048, N] = A[2048, K] * W[N, K]^T   (fp32 accumulate)
// Each wave owns an MT x NT block of 16x16 tiles: per K-step it loads
// MT A-fragments + NT B-fragments and issues MT*NT v_wmma_f32_16x16x32_bf16.
// __launch_bounds__(128,1): one wave/SIMD register budget -> no acc spills.
// #pragma unroll 1 on the K-loop: pipeline one iteration ahead only, keeping
// the live fragment set at MT+NT (not 2x) on top of MT*NT accumulators.
// ---------------------------------------------------------------------------
template <int MT, int NT>
__global__ void __launch_bounds__(128, 1)
gemm_bf16_wmma(const __bf16* __restrict__ A,
               const __bf16* __restrict__ W,
               float* __restrict__ C,
               int N, int K) {
  const int lane = threadIdx.x & 31;
  const int wid  = threadIdx.x >> 5;
  const int nblk = N / (16 * NT);
  const int mblk = SEQ / (16 * MT);
  long gw = (long)blockIdx.x * (blockDim.x >> 5) + wid;
  if (gw >= (long)nblk * mblk) return;   // wave-uniform: EXEC stays all-ones
  const int mb  = (int)(gw / nblk);
  const int nb  = (int)(gw % nblk);
  const int l15 = lane & 15;
  const int hi  = lane >> 4;

  // A fragment: lane = M row (mod 16); lanes 0-15 hold K 0-15,
  //             lanes 16-31 hold K 16-31
  const __bf16* Ap[MT];
#pragma unroll
  for (int mi = 0; mi < MT; ++mi)
    Ap[mi] = A + (long)((mb * MT + mi) * 16 + l15) * K + (hi ? 16 : 0);
  // B fragment: lane = N col (mod 16); lanes 0-15 hold K 0-15, lanes 16-31 K 16-31
  const __bf16* Wp[NT];
#pragma unroll
  for (int ni = 0; ni < NT; ++ni)
    Wp[ni] = W + (long)((nb * NT + ni) * 16 + l15) * K + (hi ? 16 : 0);

  v8f acc[MT][NT];
#pragma unroll
  for (int mi = 0; mi < MT; ++mi)
#pragma unroll
    for (int ni = 0; ni < NT; ++ni) acc[mi][ni] = v8f{};

#pragma unroll 1
  for (int k0 = 0; k0 < K; k0 += 32) {
    v16bf a[MT], b[NT];
#pragma unroll
    for (int mi = 0; mi < MT; ++mi) {
      a[mi] = *(const v16bf*)(Ap[mi] + k0);
    }
#pragma unroll
    for (int ni = 0; ni < NT; ++ni)
      b[ni] = *(const v16bf*)(Wp[ni] + k0);
#pragma unroll
    for (int mi = 0; mi < MT; ++mi)
#pragma unroll
      for (int ni = 0; ni < NT; ++ni)
        acc[mi][ni] = __builtin_amdgcn_wmma_f32_16x16x32_bf16(
            false, a[mi], false, b[ni], (short)0, acc[mi][ni], false, false);
  }

  // C/D layout: VGPR r -> M = r (lanes 0-15) or 8+r (lanes 16-31); N = lane&15
#pragma unroll
  for (int mi = 0; mi < MT; ++mi) {
    const int crow = (mb * MT + mi) * 16 + (hi ? 8 : 0);
#pragma unroll
    for (int ni = 0; ni < NT; ++ni) {
      const int ccol = (nb * NT + ni) * 16 + l15;
      float* Cp = C + (long)crow * N + ccol;
#pragma unroll
      for (int r = 0; r < 8; ++r) Cp[(long)r * N] = acc[mi][ni][r];
    }
  }
}

// ---------------------------------------------------------------------------
// Causal depthwise conv1d (k=4) + bias + SiLU on xh = xz[:, :1536]
// ---------------------------------------------------------------------------
__global__ void conv_silu(const float* __restrict__ xz,
                          const float* __restrict__ conv_w,
                          const float* __restrict__ conv_b,
                          float* __restrict__ u) {
  long i = (long)blockIdx.x * blockDim.x + threadIdx.x;
  if (i >= (long)SEQ * D_INNER) return;
  int t = (int)(i / D_INNER), c = (int)(i % D_INNER);
  float acc = conv_b[c];
#pragma unroll
  for (int j = 0; j < 4; ++j) {
    int tt = t - 3 + j;
    float xv = (tt >= 0) ? xz[(long)tt * (2 * D_INNER) + c] : 0.f;
    acc = fmaf(conv_w[c * 4 + j], xv, acc);
  }
  u[i] = acc / (1.f + __expf(-acc));   // silu
}

// ---------------------------------------------------------------------------
// x_dbl[t, 0:80] = u[t, :] @ x_proj_w^T   (K=1536, N=80; u row staged in LDS)
// ---------------------------------------------------------------------------
__global__ void xproj_kernel(const float* __restrict__ u,
                             const float* __restrict__ xpw,   // (80, 1536)
                             float* __restrict__ xdbl) {
  __shared__ float su[D_INNER];
  const int t = blockIdx.x;
  for (int k = threadIdx.x; k < D_INNER; k += blockDim.x)
    su[k] = u[(long)t * D_INNER + k];
  __syncthreads();
  if (threadIdx.x < XDBL_W) {
    const float* w = xpw + (long)threadIdx.x * D_INNER;
    float acc = 0.f;
    for (int k = 0; k < D_INNER; ++k) acc = fmaf(su[k], w[k], acc);
    xdbl[(long)t * XDBL_W + threadIdx.x] = acc;
  }
}

// ---------------------------------------------------------------------------
// dt[t, c] = softplus( xdbl[t, :48] @ dt_proj_w[c, :] + dt_proj_b[c] )
// ---------------------------------------------------------------------------
__global__ void dt_kernel(const float* __restrict__ xdbl,
                          const float* __restrict__ dtwT,   // (48, 1536)
                          const float* __restrict__ dtb,
                          float* __restrict__ dtv) {
  long i = (long)blockIdx.x * blockDim.x + threadIdx.x;
  if (i >= (long)SEQ * D_INNER) return;
  int t = (int)(i / D_INNER), c = (int)(i % D_INNER);
  const float* xr = xdbl + (long)t * XDBL_W;
  float acc = dtb[c];
#pragma unroll 8
  for (int j = 0; j < DT_RANK; ++j)
    acc = fmaf(xr[j], dtwT[(long)j * D_INNER + c], acc);
  dtv[i] = (acc > 20.f) ? acc : log1pf(__expf(acc));
}

// ---------------------------------------------------------------------------
// Selective scan: one lane per channel, 16-wide state in registers.
// B/C (shared across channels) staged in LDS in 64-timestep chunks.
// Fuses +D*u, *silu(z), and bf16 downconvert for the out_proj GEMM.
// ---------------------------------------------------------------------------
__global__ void scan_kernel(const float* __restrict__ dtv,
                            const float* __restrict__ u,
                            const float* __restrict__ xdbl,
                            const float* __restrict__ xz,
                            const float* __restrict__ A_log,
                            const float* __restrict__ Dp,
                            __bf16* __restrict__ yact) {
  const int c = blockIdx.x * blockDim.x + threadIdx.x;
  __shared__ float bc[64][32];   // [t_local][0:16]=B, [16:32]=C
  float a[D_STATE], h[D_STATE];
  float Dc = 0.f;
  if (c < D_INNER) {
    Dc = Dp[c];
#pragma unroll
    for (int s = 0; s < D_STATE; ++s) {
      a[s] = -__expf(A_log[c * D_STATE + s]);
      h[s] = 0.f;
    }
  }
  for (int t0 = 0; t0 < SEQ; t0 += 64) {
    __syncthreads();
    for (int e = threadIdx.x; e < 64 * 32; e += blockDim.x) {
      int tl = e >> 5, col = e & 31;
      bc[tl][col] = xdbl[(long)(t0 + tl) * XDBL_W + DT_RANK + col];
    }
    __syncthreads();
    if (c < D_INNER) {
      for (int tl = 0; tl < 64; ++tl) {
        const int t = t0 + tl;
        const float dt = dtv[(long)t * D_INNER + c];
        const float uv = u[(long)t * D_INNER + c];
        const float dtu = dt * uv;
        float y = 0.f;
#pragma unroll
        for (int s = 0; s < D_STATE; ++s) {
          float dA = __expf(dt * a[s]);
          h[s] = fmaf(dA, h[s], dtu * bc[tl][s]);
          y = fmaf(h[s], bc[tl][16 + s], y);
        }
        y = fmaf(Dc, uv, y);
        float zv = xz[(long)t * (2 * D_INNER) + D_INNER + c];
        float g  = zv / (1.f + __expf(-zv));
        yact[(long)t * D_INNER + c] = (__bf16)(y * g);
      }
    }
  }
}

// ---------------------------------------------------------------------------
extern "C" void kernel_launch(void* const* d_in, const int* in_sizes, int n_in,
                              void* d_out, int out_size, void* d_ws, size_t ws_size,
                              hipStream_t stream) {
  const float* x         = (const float*)d_in[0];
  const float* in_proj_w = (const float*)d_in[1];
  const float* conv_w    = (const float*)d_in[2];
  const float* conv_b    = (const float*)d_in[3];
  const float* x_proj_w  = (const float*)d_in[4];
  const float* dt_proj_w = (const float*)d_in[5];
  const float* dt_proj_b = (const float*)d_in[6];
  const float* A_log     = (const float*)d_in[7];
  const float* Dvec      = (const float*)d_in[8];
  const float* out_proj_w= (const float*)d_in[9];

  char* ws = (char*)d_ws;
  size_t off = 0;
  auto take = [&](size_t bytes) -> char* {
    char* p = ws + off;
    off = (off + bytes + 255) & ~(size_t)255;
    return p;
  };
  __bf16* xb    = (__bf16*)take((size_t)SEQ * D_MODEL * 2);
  __bf16* winb  = (__bf16*)take((size_t)2 * D_INNER * D_MODEL * 2);
  __bf16* woutb = (__bf16*)take((size_t)D_MODEL * D_INNER * 2);
  float*  xz    = (float*) take((size_t)SEQ * 2 * D_INNER * 4);
  float*  u     = (float*) take((size_t)SEQ * D_INNER * 4);
  float*  xdbl  = (float*) take((size_t)SEQ * XDBL_W * 4);
  float*  dtv   = (float*) take((size_t)SEQ * D_INNER * 4);
  float*  dtwT  = (float*) take((size_t)DT_RANK * D_INNER * 4);
  __bf16* yact  = (__bf16*)take((size_t)SEQ * D_INNER * 2);
  (void)ws_size; (void)in_sizes; (void)n_in; (void)out_size;

  // Downconverts + dt_proj_w transpose
  {
    long n1 = (long)SEQ * D_MODEL;          // x
    long n2 = (long)2 * D_INNER * D_MODEL;  // in_proj_w
    long n3 = (long)D_MODEL * D_INNER;      // out_proj_w
    cvt_f32_bf16<<<dim3((unsigned)((n1 + 255) / 256)), dim3(256), 0, stream>>>(x, xb, n1);
    cvt_f32_bf16<<<dim3((unsigned)((n2 + 255) / 256)), dim3(256), 0, stream>>>(in_proj_w, winb, n2);
    cvt_f32_bf16<<<dim3((unsigned)((n3 + 255) / 256)), dim3(256), 0, stream>>>(out_proj_w, woutb, n3);
    long nt = (long)D_INNER * DT_RANK;
    transpose_dtw<<<dim3((unsigned)((nt + 255) / 256)), dim3(256), 0, stream>>>(dt_proj_w, dtwT);
  }

  // 1) in_proj GEMM: xz = x @ in_proj_w^T   (2048 x 3072, K=768), 2x4 blocked
  {
    const int waves = (SEQ / 32) * ((2 * D_INNER) / 64);   // 64 * 48 = 3072
    gemm_bf16_wmma<2, 4><<<dim3(waves / 4), dim3(128), 0, stream>>>(
        xb, winb, xz, 2 * D_INNER, D_MODEL);
  }

  // 2) depthwise causal conv + SiLU -> u
  {
    long n = (long)SEQ * D_INNER;
    conv_silu<<<dim3((unsigned)((n + 255) / 256)), dim3(256), 0, stream>>>(xz, conv_w, conv_b, u);
  }

  // 3) x_proj: xdbl = u @ x_proj_w^T   (2048 x 80)
  xproj_kernel<<<dim3(SEQ), dim3(128), 0, stream>>>(u, x_proj_w, xdbl);

  // 4) dt = softplus(xdbl[:, :48] @ dt_proj_w^T + b)
  {
    long n = (long)SEQ * D_INNER;
    dt_kernel<<<dim3((unsigned)((n + 255) / 256)), dim3(256), 0, stream>>>(xdbl, dtwT, dt_proj_b, dtv);
  }

  // 5) selective scan + D*u + silu(z) gating -> yact (bf16)
  scan_kernel<<<dim3(D_INNER / 256), dim3(256), 0, stream>>>(
      dtv, u, xdbl, xz, A_log, Dvec, yact);

  // 6) out_proj GEMM: out = yact @ out_proj_w^T (2048 x 768, K=1536), 2x2 blocked
  {
    const int waves = (SEQ / 32) * (D_MODEL / 32);   // 64 * 24 = 1536
    gemm_bf16_wmma<2, 2><<<dim3(waves / 4), dim3(128), 0, stream>>>(
        yact, woutb, (float*)d_out, D_MODEL, D_INNER);
  }
}
